// BeitSelfAttention_75677323755713
// MI455X (gfx1250) — compile-verified, matrix operand order checked by reading, not run on hardware
//
#include <hip/hip_runtime.h>
#include <cstdint>

typedef __attribute__((ext_vector_type(16))) __bf16       v16bf;
typedef __attribute__((ext_vector_type(8)))  float        v8f;
typedef __attribute__((ext_vector_type(4)))  unsigned int v4u;
typedef __attribute__((ext_vector_type(4)))  int          v4i;

union Frag {
  v16bf bf;
  v4u   u[2];
};

#define BATCH  32
#define NTOK   577
#define DMODEL 768
#define NHEAD  12
#define HD     64
#define NP     608            // padded key-stride for V^T rows (16B-aligned, covers 19*32 keys)
#define MROWS  (BATCH * NTOK) // 18464 = 1154 * 16 exactly

#if defined(__gfx1250__) && __has_builtin(__builtin_amdgcn_global_load_async_to_lds_b128)
#define USE_ASYNC 1
#else
#define USE_ASYNC 0
#endif

__device__ __forceinline__ void async_copy16(const __bf16* g, __bf16* l) {
#if USE_ASYNC
  __builtin_amdgcn_global_load_async_to_lds_b128(
      (__attribute__((address_space(1))) v4i*)(v4i*)g,
      (__attribute__((address_space(3))) v4i*)(v4i*)l, 0, 0);
#else
  *(v4u*)l = *(const v4u*)g;
#endif
}

__device__ __forceinline__ void async_wait_all() {
#if USE_ASYNC
#if __has_builtin(__builtin_amdgcn_s_wait_asynccnt)
  __builtin_amdgcn_s_wait_asynccnt(0);
#else
  asm volatile("s_wait_asynccnt 0" ::: "memory");
#endif
#endif
}

__device__ __forceinline__ v8f vzero8() {
  v8f v;
#pragma unroll
  for (int i = 0; i < 8; ++i) v[i] = 0.0f;
  return v;
}

// ---------------- f32 -> bf16 convert ----------------
__global__ __launch_bounds__(256) void cvt_bf16(const float* __restrict__ src,
                                                __bf16* __restrict__ dst, int n) {
  int i = blockIdx.x * 256 + threadIdx.x;
  if (i < n) dst[i] = (__bf16)src[i];
}

// ---------------- bias gather-expand: [H][N][N] f32 ----------------
__global__ __launch_bounds__(256) void bias_expand(const float* __restrict__ table,
                                                   const int* __restrict__ idx,
                                                   float* __restrict__ bias) {
  int qk = blockIdx.x * 256 + threadIdx.x;
  if (qk >= NTOK * NTOK) return;
  int h = blockIdx.y;
  bias[(size_t)h * NTOK * NTOK + qk] = table[(size_t)idx[qk] * NHEAD + h];
}

// ---------------- fused QKV projection GEMM (bf16 WMMA, f32 accum) ----------------
// Y = X @ W^T (+bias). W stored row-major: row c of W == column c of W^T == B-fragment n=c.
// Each wave computes a 16x64 tile (one head's hd-block) of one projection.
__global__ __launch_bounds__(128) void qkv_gemm(const __bf16* __restrict__ X,
                                                const __bf16* __restrict__ W,   // [3][768][768]
                                                const float* __restrict__ qb,
                                                const float* __restrict__ vb,
                                                __bf16* __restrict__ Q,    // [B,H,N,HD], pre-scaled 1/8
                                                __bf16* __restrict__ Kc,   // [B,H,N,HD]
                                                __bf16* __restrict__ VT) { // [B,H,HD,NP]
  const int lane = threadIdx.x & 31;
  const int wv   = threadIdx.x >> 5;
  const int ml   = lane & 15;
  const int hf   = lane >> 4;

  const int wt = blockIdx.x * 4 + wv;   // wave tile id, 0..41543
  const int mt = wt / 36;               // 16-row block of M = B*N
  const int ct = wt % 36;               // 64-col block across {q,k,v} x 12 heads
  const int p  = ct / NHEAD;            // 0=q, 1=k, 2=v
  const int h  = ct % NHEAD;

  const __bf16* xrow  = X + (size_t)(mt * 16 + ml) * DMODEL;
  const __bf16* wbase = W + (size_t)p * DMODEL * DMODEL;

  v8f acc[4];
#pragma unroll
  for (int j = 0; j < 4; ++j) acc[j] = vzero8();

  for (int kc = 0; kc < DMODEL / 32; ++kc) {
    Frag a;
    a.u[0] = *(const v4u*)(xrow + kc * 32 + 8 * hf);
    a.u[1] = *(const v4u*)(xrow + kc * 32 + 16 + 8 * hf);
#pragma unroll
    for (int j = 0; j < 4; ++j) {
      const __bf16* wrow = wbase + (size_t)(h * HD + j * 16 + ml) * DMODEL;
      Frag b;
      b.u[0] = *(const v4u*)(wrow + kc * 32 + 8 * hf);
      b.u[1] = *(const v4u*)(wrow + kc * 32 + 16 + 8 * hf);
      acc[j] = __builtin_amdgcn_wmma_f32_16x16x32_bf16(false, a.bf, false, b.bf,
                                                       (short)0, acc[j], false, false);
    }
  }

#pragma unroll
  for (int j = 0; j < 4; ++j) {
#pragma unroll
    for (int r = 0; r < 8; ++r) {
      int gm  = mt * 16 + r + 8 * hf;   // global M row (always < MROWS)
      int bb  = gm / NTOK;
      int tok = gm % NTOK;
      int hd  = j * 16 + ml;
      int c   = h * HD + hd;
      float x = acc[j][r];
      if (p == 0) {
        Q[(((size_t)bb * NHEAD + h) * NTOK + tok) * HD + hd] = (__bf16)((x + qb[c]) * 0.125f);
      } else if (p == 1) {
        Kc[(((size_t)bb * NHEAD + h) * NTOK + tok) * HD + hd] = (__bf16)x;
      } else {
        VT[(((size_t)bb * NHEAD + h) * HD + hd) * NP + tok] = (__bf16)(x + vb[c]);
      }
    }
  }
}

// ---------------- flash attention: online softmax + WMMA + async K/V staging ----------------
// Block = 4 waves; each wave owns a 16-row q tile (64 q rows / block). All 4 waves walk the
// same key tiles, so K (32x64 bf16) and V^T (64x32 bf16) are staged once per block into LDS
// with ASYNCcnt-tracked global_load_async_to_lds_b128.
__global__ __launch_bounds__(128) void attn(const __bf16* __restrict__ Q,
                                            const __bf16* __restrict__ K,
                                            const __bf16* __restrict__ VT,
                                            const float* __restrict__ bias,
                                            float* __restrict__ out) {
  __shared__ __bf16 Kl[32][64];        // key tile, row-major per key     (4 KB)
  __shared__ __bf16 Vl[64][32];        // V^T tile, row-major per hd      (4 KB)
  __shared__ __bf16 plds[4][16][48];   // per-wave P staging (96B stride) (6 KB)

  const int t    = threadIdx.x;
  const int lane = t & 31;
  const int wv   = t >> 5;
  const int ml   = lane & 15;
  const int hf   = lane >> 4;
  const int bh   = blockIdx.y;             // b*12 + h
  const int b    = bh / NHEAD;
  const int h    = bh % NHEAD;
  const int q0   = blockIdx.x * 64 + wv * 16;   // may exceed NTOK; clamped below, stores guarded

  // Q fragments: 16 rows x 64 k (two 16x32 A-fragments)
  int qr_l = q0 + ml; if (qr_l > NTOK - 1) qr_l = NTOK - 1;
  const __bf16* qrow = Q + ((size_t)bh * NTOK + qr_l) * HD;
  Frag aq[2];
#pragma unroll
  for (int c = 0; c < 2; ++c) {
    aq[c].u[0] = *(const v4u*)(qrow + c * 32 + 8 * hf);
    aq[c].u[1] = *(const v4u*)(qrow + c * 32 + 16 + 8 * hf);
  }

  v8f o[4];
#pragma unroll
  for (int j = 0; j < 4; ++j) o[j] = vzero8();
  float mrun[8], lrun[8];
#pragma unroll
  for (int r = 0; r < 8; ++r) { mrun[r] = -1e30f; lrun[r] = 0.0f; }

  const float* bias_h = bias + (size_t)h * NTOK * NTOK;
  const __bf16* kbase = K + (size_t)bh * NTOK * HD;
  const __bf16* vbase = VT + (size_t)bh * HD * NP;

  for (int kt = 0; kt < (NTOK + 31) / 32; ++kt) {
    const int key0 = kt * 32;

    // ---- cooperative async staging of K tile and V^T tile into LDS ----
#pragma unroll
    for (int L = 0; L < 2; ++L) {           // K: 256 x 16B chunks over 128 threads
      int li  = t + L * 128;
      int row = li >> 3, cb = (li & 7) * 8; // 32 rows x 8 chunks
      int key = key0 + row; if (key > NTOK - 1) key = NTOK - 1;
      async_copy16(kbase + (size_t)key * HD + cb, &Kl[row][cb]);
    }
#pragma unroll
    for (int L = 0; L < 2; ++L) {           // V^T: 256 x 16B chunks over 128 threads
      int li  = t + L * 128;
      int row = li >> 2, cb = (li & 3) * 8; // 64 rows x 4 chunks
      async_copy16(vbase + (size_t)row * NP + key0 + cb, &Vl[row][cb]);
    }
    async_wait_all();
    __syncthreads();

    // prefetch next tile's bias row segment while computing this one
    {
      int qr_p = q0 + 8 * hf; if (qr_p > NTOK - 1) qr_p = NTOK - 1;
      __builtin_prefetch(bias_h + (size_t)qr_p * NTOK + key0 + 32, 0, 0);
    }

    // ---- S = Q * K^T (two 16x16 tiles covering 32 keys) ----
    v8f s[2];
#pragma unroll
    for (int st = 0; st < 2; ++st) {
      s[st] = vzero8();
      const __bf16* krow = &Kl[st * 16 + ml][0];
#pragma unroll
      for (int c = 0; c < 2; ++c) {
        Frag bk;
        bk.u[0] = *(const v4u*)(krow + c * 32 + 8 * hf);
        bk.u[1] = *(const v4u*)(krow + c * 32 + 16 + 8 * hf);
        s[st] = __builtin_amdgcn_wmma_f32_16x16x32_bf16(false, aq[c].bf, false, bk.bf,
                                                        (short)0, s[st], false, false);
      }
    }

    // ---- bias + mask + online softmax (row lives across one 16-lane half-group) ----
#pragma unroll
    for (int r = 0; r < 8; ++r) {
      int qr = q0 + r + 8 * hf; if (qr > NTOK - 1) qr = NTOK - 1;
      const float* brow = bias_h + (size_t)qr * NTOK;
#pragma unroll
      for (int st = 0; st < 2; ++st) {
        int col = key0 + st * 16 + ml;
        s[st][r] = (col < NTOK) ? (s[st][r] + brow[col]) : -1e30f;
      }
      float rm = fmaxf(s[0][r], s[1][r]);
#pragma unroll
      for (int off = 8; off >= 1; off >>= 1)
        rm = fmaxf(rm, __shfl_xor(rm, off, 32));
      float nm    = fmaxf(mrun[r], rm);
      float alpha = __expf(mrun[r] - nm);
      float p0    = __expf(s[0][r] - nm);
      float p1    = __expf(s[1][r] - nm);
      float rs    = p0 + p1;
#pragma unroll
      for (int off = 8; off >= 1; off >>= 1)
        rs += __shfl_xor(rs, off, 32);
      lrun[r] = lrun[r] * alpha + rs;
      mrun[r] = nm;
#pragma unroll
      for (int j = 0; j < 4; ++j) o[j][r] *= alpha;
      // stash P (C-layout) into per-wave LDS region for A-fragment re-read
      int m = r + 8 * hf;
      plds[wv][m][ml]      = (__bf16)p0;
      plds[wv][m][16 + ml] = (__bf16)p1;
    }

    // re-read P as 16x32 A-fragment (same-wave LDS ops are in-order)
    Frag ap;
    ap.u[0] = *(const v4u*)&plds[wv][ml][8 * hf];
    ap.u[1] = *(const v4u*)&plds[wv][ml][16 + 8 * hf];

    // ---- O += P * V (V^T rows are contiguous in key) ----
#pragma unroll
    for (int j = 0; j < 4; ++j) {
      const __bf16* vrow = &Vl[j * 16 + ml][0];
      Frag bv;
      bv.u[0] = *(const v4u*)(vrow + 8 * hf);
      bv.u[1] = *(const v4u*)(vrow + 16 + 8 * hf);
      o[j] = __builtin_amdgcn_wmma_f32_16x16x32_bf16(false, ap.bf, false, bv.bf,
                                                     (short)0, o[j], false, false);
    }

    __syncthreads();   // Kl/Vl fully consumed before next tile overwrites them
  }

  // epilogue: divide by softmax denominator, write [B,N,D] f32
#pragma unroll
  for (int j = 0; j < 4; ++j) {
#pragma unroll
    for (int r = 0; r < 8; ++r) {
      int qr = q0 + r + 8 * hf;
      if (qr < NTOK)
        out[((size_t)b * NTOK + qr) * DMODEL + h * HD + j * 16 + ml] = o[j][r] / lrun[r];
    }
  }
}

extern "C" void kernel_launch(void* const* d_in, const int* in_sizes, int n_in,
                              void* d_out, int out_size, void* d_ws, size_t ws_size,
                              hipStream_t stream) {
  (void)in_sizes; (void)n_in; (void)out_size; (void)ws_size;
  const float* X     = (const float*)d_in[0];
  const float* qw    = (const float*)d_in[1];
  const float* qb    = (const float*)d_in[2];
  const float* kw    = (const float*)d_in[3];
  const float* vw    = (const float*)d_in[4];
  const float* vb    = (const float*)d_in[5];
  const float* table = (const float*)d_in[6];
  const int*   idx   = (const int*)d_in[7];
  float* out = (float*)d_out;

  char* ws = (char*)d_ws;
  const size_t nX = (size_t)MROWS * DMODEL;   // 14,180,352 elems
  const size_t nW = (size_t)DMODEL * DMODEL;  //    589,824 elems
  __bf16* Xbf  = (__bf16*)(ws);
  __bf16* Wbf  = (__bf16*)(ws + 2 * nX);
  __bf16* Qbf  = (__bf16*)(ws + 2 * nX + 2 * 3 * nW);
  __bf16* Kbf  = (__bf16*)((char*)Qbf + 2 * nX);
  __bf16* VTbf = (__bf16*)((char*)Kbf + 2 * nX);
  float*  biasf = (float*)((char*)VTbf + 2 * (size_t)BATCH * NHEAD * HD * NP);

  cvt_bf16<<<(int)((nX + 255) / 256), 256, 0, stream>>>(X, Xbf, (int)nX);
  cvt_bf16<<<(int)((nW + 255) / 256), 256, 0, stream>>>(qw, Wbf + 0 * nW, (int)nW);
  cvt_bf16<<<(int)((nW + 255) / 256), 256, 0, stream>>>(kw, Wbf + 1 * nW, (int)nW);
  cvt_bf16<<<(int)((nW + 255) / 256), 256, 0, stream>>>(vw, Wbf + 2 * nW, (int)nW);
  bias_expand<<<dim3((NTOK * NTOK + 255) / 256, NHEAD), 256, 0, stream>>>(table, idx, biasf);
  qkv_gemm<<<(MROWS / 16) * 36 / 4, 128, 0, stream>>>(Xbf, Wbf, qb, vb, Qbf, Kbf, VTbf);
  attn<<<dim3((NTOK + 63) / 64, BATCH * NHEAD), 128, 0, stream>>>(Qbf, Kbf, VTbf, biasf, out);
}
